// HGATImputer_17901423690206
// MI455X (gfx1250) — compile-verified, matrix operand order checked by reading, not run on hardware
//
#include <hip/hip_runtime.h>
#include <hip/hip_bf16.h>
#include <math.h>

typedef __attribute__((ext_vector_type(16))) __bf16 v16bf;
typedef __attribute__((ext_vector_type(8)))  __bf16 v8bf;
typedef __attribute__((ext_vector_type(2)))  __bf16 v2bf;
typedef __attribute__((ext_vector_type(8)))  float  v8f;

constexpr int BBATCH = 8;
constexpr int INC    = 64;
constexpr int NN     = 4000;   // nodes
constexpr int EE     = 1000;   // edges
constexpr int KK     = 256;    // feature dim
constexpr int DD     = 192;    // 3*IN

constexpr int BM = 64, BN = 128, BK = 32;
constexpr int LDT = 40;        // padded LDS row pitch in bf16 (80B, 16B-aligned)

__device__ __forceinline__ float lrelu(float v) { return v > 0.f ? v : 0.2f * v; }
__device__ __forceinline__ __bf16 f2bf(float v) { return (__bf16)v; }

// ---------------- GEMM loader / epilogue functors ----------------

struct ALoadXMH {                    // A[n,d] = concat(x,m,hidden)[b,d,n]
  static constexpr bool kContig = false;   // contiguous along m (=n)
  const float *x, *m, *h;
  __device__ float operator()(int b, int mm, int kk) const {
    const float* s = (kk < INC) ? x : (kk < 2 * INC ? m : h);
    return s[((size_t)b * INC + (kk & (INC - 1))) * NN + mm];
  }
};
struct ALoadInc {                    // A[e,n] = inc[n,e]
  static constexpr bool kContig = false;   // contiguous along m (=e)
  const float* inc;
  __device__ float operator()(int b, int mm, int kk) const {
    (void)b; return inc[(size_t)kk * EE + mm];
  }
};
struct ALoadW2T {                    // A[f,e] = W2[e,f]
  static constexpr bool kContig = false;
  const float* w2;
  __device__ float operator()(int b, int mm, int kk) const {
    (void)b; return w2[(size_t)kk * EE + mm];
  }
};
struct ALoadWT {                     // A[n,e] = wT[b,e,n]  (bf16)
  static constexpr bool kContig = false;
  const __bf16* wt;
  __device__ float operator()(int b, int mm, int kk) const {
    return (float)wt[((size_t)b * EE + kk) * NN + mm];
  }
};
struct BRowMajor {                   // B[k,n] row-major f32 (+batch stride)
  const float* p; int ld; size_t bstride;
  __device__ float operator()(int b, int kk, int nn) const {
    return p[(size_t)b * bstride + (size_t)kk * ld + nn];
  }
};
struct EpFeat {
  float* out; const float* bias;
  __device__ void operator()(int b, int mm, int nn, float v) const {
    out[((size_t)b * NN + mm) * KK + nn] = v + bias[nn];
  }
};
struct EpEdgeM {
  float* out; const float* invdeg;
  __device__ void operator()(int b, int mm, int nn, float v) const {
    out[((size_t)b * EE + mm) * KK + nn] = v * invdeg[mm];
  }
};
struct EpEdge2 {
  float* ws; float* out2;
  __device__ void operator()(int b, int mm, int nn, float v) const {
    size_t i = ((size_t)b * EE + mm) * KK + nn;
    ws[i] = v; out2[i] = v;
  }
};
struct EpNodeT {                     // node.transpose(0,2,1)
  float* out;
  __device__ void operator()(int b, int mm, int nn, float v) const {
    out[((size_t)b * KK + nn) * NN + mm] = v;
  }
};

// -------- WMMA GEMM: D = A(MxK)*B(KxN), fp32 acc, bf16 mma, pipelined --------
// LDS: A as [m][k] (row pitch LDT), B transposed as [n][k] (row pitch LDT)
// -> each lane's 16-element fragment is two contiguous 16B runs: 2x ds_load_b128
// All fragments preloaded before the 4 WMMAs so LDS latency overlaps issue.

template <class AL, class BL, class EP>
__global__ __launch_bounds__(256) void gemm_wmma(AL al, BL bl, EP ep,
                                                 int M, int N, int K) {
  __shared__ __bf16 sA[BM * LDT];    // 5  KB
  __shared__ __bf16 sB[BN * LDT];    // 10 KB
  const int tid  = threadIdx.x;
  const int lane = tid & 31, wave = tid >> 5;
  const int wm = wave & 3, wn = wave >> 2;        // 4x2 wave grid -> 64x128
  const int m0 = blockIdx.y * BM, n0 = blockIdx.x * BN;
  const int b  = blockIdx.z;

  float ra[8];    // 4 k-pairs of A per thread
  float rb[16];   // 8 k-pairs of B per thread

  auto loadA = [&](int kt) {
#pragma unroll
    for (int j = 0; j < 4; ++j) {
      int i = tid + j * 256;
      int mloc, k2;
      if constexpr (AL::kContig) { k2 = (i & 15) * 2; mloc = i >> 4; }
      else                       { mloc = i & 63;     k2 = (i >> 6) * 2; }
      int gm = m0 + mloc, gk = kt + k2;
      ra[2 * j]     = (gm < M && gk     < K) ? al(b, gm, gk)     : 0.f;
      ra[2 * j + 1] = (gm < M && gk + 1 < K) ? al(b, gm, gk + 1) : 0.f;
    }
  };
  auto loadB = [&](int kt) {
#pragma unroll
    for (int j = 0; j < 8; ++j) {
      int i = tid + j * 256;
      int n = i & 127, k2 = (i >> 7) * 2;
      int gn = n0 + n, gk = kt + k2;
      rb[2 * j]     = (gk     < K && gn < N) ? bl(b, gk,     gn) : 0.f;
      rb[2 * j + 1] = (gk + 1 < K && gn < N) ? bl(b, gk + 1, gn) : 0.f;
    }
  };
  auto storeAB = [&]() {
#pragma unroll
    for (int j = 0; j < 4; ++j) {
      int i = tid + j * 256;
      int mloc, k2;
      if constexpr (AL::kContig) { k2 = (i & 15) * 2; mloc = i >> 4; }
      else                       { mloc = i & 63;     k2 = (i >> 6) * 2; }
      v2bf p = { f2bf(ra[2 * j]), f2bf(ra[2 * j + 1]) };
      *reinterpret_cast<v2bf*>(&sA[mloc * LDT + k2]) = p;
    }
#pragma unroll
    for (int j = 0; j < 8; ++j) {
      int i = tid + j * 256;
      int n = i & 127, k2 = (i >> 7) * 2;
      v2bf p = { f2bf(rb[2 * j]), f2bf(rb[2 * j + 1]) };
      *reinterpret_cast<v2bf*>(&sB[n * LDT + k2]) = p;
    }
  };

  v8f zero = {0.f, 0.f, 0.f, 0.f, 0.f, 0.f, 0.f, 0.f};
  v8f acc[4];
#pragma unroll
  for (int t = 0; t < 4; ++t) acc[t] = zero;

  const int rowA = wm * 16 + (lane & 15);
  const int kb   = (lane >> 4) << 3;          // 0 or 8
  const int colB = wn * 64 + (lane & 15);

  const int steps = (K + BK - 1) / BK;
  loadA(0); loadB(0);
  for (int s = 0; s < steps; ++s) {
    __syncthreads();                 // previous compute finished, buffers free
    storeAB();
    __syncthreads();                 // tiles visible
    if (s + 1 < steps) { loadA((s + 1) * BK); loadB((s + 1) * BK); }  // prefetch

    // preload ALL fragments first (10x ds_load_b128), then issue the WMMAs
    const __bf16* pa = &sA[rowA * LDT + kb];
    v8bf alo = *reinterpret_cast<const v8bf*>(pa);        // K = kb..kb+7
    v8bf ahi = *reinterpret_cast<const v8bf*>(pa + 16);   // K = 16+kb..16+kb+7
    v16bf a = __builtin_shufflevector(alo, ahi, 0, 1, 2, 3, 4, 5, 6, 7,
                                      8, 9, 10, 11, 12, 13, 14, 15);
    v16bf bfr[4];
#pragma unroll
    for (int t = 0; t < 4; ++t) {
      const __bf16* pb = &sB[(colB + t * 16) * LDT + kb];
      v8bf blo = *reinterpret_cast<const v8bf*>(pb);
      v8bf bhi = *reinterpret_cast<const v8bf*>(pb + 16);
      bfr[t] = __builtin_shufflevector(blo, bhi, 0, 1, 2, 3, 4, 5, 6, 7,
                                       8, 9, 10, 11, 12, 13, 14, 15);
    }
#pragma unroll
    for (int t = 0; t < 4; ++t) {
      acc[t] = __builtin_amdgcn_wmma_f32_16x16x32_bf16(
          false, a, false, bfr[t], (short)0, acc[t], false, false);
    }
  }

  // C/D layout: lane<16 -> M=r, lane>=16 -> M=8+r; N = lane&15
  const int mbase = m0 + wm * 16 + (lane >> 4) * 8;
  const int nl = lane & 15;
#pragma unroll
  for (int t = 0; t < 4; ++t) {
    const int gn = n0 + wn * 64 + t * 16 + nl;
#pragma unroll
    for (int r = 0; r < 8; ++r) {
      const int gm = mbase + r;
      if (gm < M && gn < N) ep(b, gm, gn, acc[t][r]);
    }
  }
}

// ---------------- small kernels ----------------

__global__ __launch_bounds__(256) void k_deg(const float* inc, float* invdeg) {
  int e = blockIdx.x * 256 + threadIdx.x;
  if (e >= EE) return;
  float s = 0.f;
  for (int n = 0; n < NN; ++n) s += inc[(size_t)n * EE + e];  // coalesced over e
  invdeg[e] = 1.f / s;                                         // deg >= 1 (diag)
}

__global__ __launch_bounds__(256) void k_transpose(const float* in, float* out) {
  __shared__ float tile[32][33];
  int n0 = blockIdx.x * 32, e0 = blockIdx.y * 32;
  int tx = threadIdx.x & 31, ty = threadIdx.x >> 5;
  for (int j = 0; j < 32; j += 8) {
    int n = n0 + ty + j, e = e0 + tx;
    tile[ty + j][tx] = (n < NN && e < EE) ? in[(size_t)n * EE + e] : 0.f;
  }
  __syncthreads();
  for (int j = 0; j < 32; j += 8) {
    int e = e0 + ty + j, n = n0 + tx;
    if (e < EE && n < NN) out[(size_t)e * NN + n] = tile[tx][ty + j];
  }
}

// one wave per row: fax[b,n] = dot(feat[b,n,:], ax)
__global__ __launch_bounds__(256) void k_fax(const float* feat, const float* a,
                                             float* fax) {
  int w = (blockIdx.x * 256 + threadIdx.x) >> 5;
  int lane = threadIdx.x & 31;
  if (w >= BBATCH * NN) return;
  const float* row = feat + (size_t)w * KK;
  float s = 0.f;
  for (int i = lane; i < KK; i += 32) s += row[i] * a[i];
  for (int o = 16; o; o >>= 1) s += __shfl_xor(s, o, 32);
  if (lane == 0) fax[w] = s;
}

// one wave per row: eae[b,e]=dot(edge,ae); te[b,e]=lrelu(dot(edge,a2)+pri*a2[K])
__global__ __launch_bounds__(256) void k_edgevec(const float* edge,
                                                 const float* a,
                                                 const float* a2,
                                                 const float* pri_e,
                                                 float* eae, float* te) {
  int w = (blockIdx.x * 256 + threadIdx.x) >> 5;
  int lane = threadIdx.x & 31;
  if (w >= BBATCH * EE) return;
  const float* row = edge + (size_t)w * KK;
  float s_ae = 0.f, s_a2 = 0.f;
  for (int i = lane; i < KK; i += 32) {
    float v = row[i];
    s_ae += v * a[KK + i];
    s_a2 += v * a2[i];
  }
  for (int o = 16; o; o >>= 1) {
    s_ae += __shfl_xor(s_ae, o, 32);
    s_a2 += __shfl_xor(s_a2, o, 32);
  }
  if (lane == 0) {
    eae[w] = s_ae;
    te[w] = lrelu(s_a2 + pri_e[(size_t)w * 2 + 1] * a2[KK]);
  }
}

// masked softmax over n for each (b,e); write w as bf16 in (b,e,n) layout
__global__ __launch_bounds__(256) void k_softmax(const float* incT,
                                                 const float* fax,
                                                 const float* eae,
                                                 const float* te,
                                                 const float* a3,
                                                 __bf16* wT) {
  const int e = blockIdx.x, b = blockIdx.y;
  const int tid = threadIdx.x;
  const float a30 = a3[0], a31 = a3[1];
  const float tev = te[b * EE + e], eaev = eae[b * EE + e];
  const float* incc = incT + (size_t)e * NN;
  const float* faxb = fax + (size_t)b * NN;
  __bf16* wrow = wT + ((size_t)b * EE + e) * NN;
  __shared__ float sh[8];
  __shared__ float bc;

  // pass 1: max
  float mx = -3.4e38f;
  for (int n = tid; n < NN; n += 256)
    if (incc[n] > 0.f)
      mx = fmaxf(mx, lrelu(tev * a30 + lrelu(faxb[n] + eaev) * a31));
  for (int o = 16; o; o >>= 1) mx = fmaxf(mx, __shfl_xor(mx, o, 32));
  if ((tid & 31) == 0) sh[tid >> 5] = mx;
  __syncthreads();
  if (tid == 0) {
    float v = sh[0];
    for (int i = 1; i < 8; ++i) v = fmaxf(v, sh[i]);
    bc = v;
  }
  __syncthreads();
  mx = bc;

  // pass 2: sum of exp
  float sm = 0.f;
  for (int n = tid; n < NN; n += 256)
    if (incc[n] > 0.f)
      sm += expf(lrelu(tev * a30 + lrelu(faxb[n] + eaev) * a31) - mx);
  for (int o = 16; o; o >>= 1) sm += __shfl_xor(sm, o, 32);
  __syncthreads();
  if ((tid & 31) == 0) sh[tid >> 5] = sm;
  __syncthreads();
  if (tid == 0) {
    float v = 0.f;
    for (int i = 0; i < 8; ++i) v += sh[i];
    bc = 1.f / v;
  }
  __syncthreads();
  const float inv = bc;

  // pass 3: normalized weights (masked -> 0)
  for (int n = tid; n < NN; n += 256) {
    float w = 0.f;
    if (incc[n] > 0.f)
      w = expf(lrelu(tev * a30 + lrelu(faxb[n] + eaev) * a31) - mx) * inv;
    wrow[n] = f2bf(w);
  }
}

// ---------------- launcher ----------------

extern "C" void kernel_launch(void* const* d_in, const int* in_sizes, int n_in,
                              void* d_out, int out_size, void* d_ws,
                              size_t ws_size, hipStream_t stream) {
  (void)in_sizes; (void)n_in; (void)out_size; (void)ws_size;
  const float* x      = (const float*)d_in[0];
  const float* m      = (const float*)d_in[1];
  const float* hidden = (const float*)d_in[2];
  const float* pri_e  = (const float*)d_in[3];
  // d_in[4] = pri_n (unused by reference)
  const float* inc    = (const float*)d_in[5];
  const float* weight = (const float*)d_in[6];
  const float* bias   = (const float*)d_in[7];
  const float* w2     = (const float*)d_in[8];
  const float* a      = (const float*)d_in[9];
  const float* a2     = (const float*)d_in[10];
  const float* a3     = (const float*)d_in[11];

  char* ws = (char*)d_ws;
  size_t off = 0;
  float*  featF  = (float*)(ws + off); off += (size_t)BBATCH * NN * KK * 4;
  float*  edgeM  = (float*)(ws + off); off += (size_t)BBATCH * EE * KK * 4;
  float*  edge2  = (float*)(ws + off); off += (size_t)BBATCH * EE * KK * 4;
  float*  incT   = (float*)(ws + off); off += (size_t)EE * NN * 4;
  __bf16* wT     = (__bf16*)(ws + off); off += (size_t)BBATCH * EE * NN * 2;
  float*  invdeg = (float*)(ws + off); off += 4096;
  float*  fax    = (float*)(ws + off); off += (size_t)BBATCH * NN * 4;
  float*  eae    = (float*)(ws + off); off += (size_t)BBATCH * EE * 4;
  float*  te     = (float*)(ws + off); off += (size_t)BBATCH * EE * 4;

  float* outNode = (float*)d_out;
  float* outEdge = (float*)d_out + (size_t)BBATCH * KK * NN;

  // degree + transposed incidence
  k_deg<<<(EE + 255) / 256, 256, 0, stream>>>(inc, invdeg);
  k_transpose<<<dim3((NN + 31) / 32, (EE + 31) / 32), 256, 0, stream>>>(inc, incT);

  // GEMM 1: feat = xin^T * W + bias       (M=NN, N=KK, K=DD)
  gemm_wmma<<<dim3(KK / BN, (NN + BM - 1) / BM, BBATCH), 256, 0, stream>>>(
      ALoadXMH{x, m, hidden}, BRowMajor{weight, KK, 0},
      EpFeat{featF, bias}, NN, KK, DD);

  // GEMM 2: edgeM = (inc^T * feat)/deg    (M=EE, N=KK, K=NN)
  gemm_wmma<<<dim3(KK / BN, (EE + BM - 1) / BM, BBATCH), 256, 0, stream>>>(
      ALoadInc{inc}, BRowMajor{featF, KK, (size_t)NN * KK},
      EpEdgeM{edgeM, invdeg}, EE, KK, NN);

  // GEMM 3: edge2 = W2^T * edgeM          (M=EE, N=KK, K=EE), dual store
  gemm_wmma<<<dim3(KK / BN, (EE + BM - 1) / BM, BBATCH), 256, 0, stream>>>(
      ALoadW2T{w2}, BRowMajor{edgeM, KK, (size_t)EE * KK},
      EpEdge2{edge2, outEdge}, EE, KK, EE);

  // per-row dot products
  k_fax<<<(BBATCH * NN * 32 + 255) / 256, 256, 0, stream>>>(featF, a, fax);
  k_edgevec<<<(BBATCH * EE * 32 + 255) / 256, 256, 0, stream>>>(edge2, a, a2,
                                                                pri_e, eae, te);

  // masked softmax over n -> w (bf16, (b,e,n))
  k_softmax<<<dim3(EE, BBATCH), 256, 0, stream>>>(incT, fax, eae, te, a3, wT);

  // GEMM 4: node = w * edge2, stored transposed   (M=NN, N=KK, K=EE)
  gemm_wmma<<<dim3(KK / BN, (NN + BM - 1) / BM, BBATCH), 256, 0, stream>>>(
      ALoadWT{wT}, BRowMajor{edge2, KK, (size_t)EE * KK},
      EpNodeT{outNode}, NN, KK, EE);
}
